// BasicLayersEncoderGPT2_34703335752348
// MI455X (gfx1250) — matrix operand order repaired and checked by reading, not compiled, not run on hardware
//
#include <hip/hip_runtime.h>
#include <hip/hip_bf16.h>
#include <math.h>

// ---------------------------------------------------------------------------
// Types for CDNA5 WMMA
// ---------------------------------------------------------------------------
typedef __attribute__((ext_vector_type(16))) __bf16 v16bf;
typedef __attribute__((ext_vector_type(8)))  __bf16 v8bf;
typedef __attribute__((ext_vector_type(8)))  float  v8f;

#define EOS_TOK 50256

__device__ __forceinline__ __bf16 to_bf16(float f) { return (__bf16)f; }

// ---------------------------------------------------------------------------
// 1) Segment / relative-index scan
// ---------------------------------------------------------------------------
__global__ void seg_scan_kernel(const int* __restrict__ ids,
                                int* __restrict__ seg, int* __restrict__ rel,
                                int L) {
  if (threadIdx.x != 0) return;
  int b = blockIdx.x;
  int last = 0, s = 0;
  for (int t = 0; t < L; ++t) {
    int id = ids[b * L + t];
    int v = (id == EOS_TOK) ? t : 0;
    if (v > last) last = v;
    rel[b * L + t] = t - last;
    s += (id == EOS_TOK) ? 1 : 0;
    seg[b * L + t] = s;
  }
}

// ---------------------------------------------------------------------------
// 2) Embedding: x = tok_emb[id]*sqrt(D) + pos_emb[rel]
// ---------------------------------------------------------------------------
__global__ void embed_kernel(const int* __restrict__ ids,
                             const int* __restrict__ rel,
                             const float* __restrict__ tok,
                             const float* __restrict__ pos,
                             float* __restrict__ x, int D) {
  int m = blockIdx.x;
  int id = ids[m];
  int r  = rel[m];
  const float sc = 27.712812921102035f;  // sqrt(768)
  for (int d = threadIdx.x; d < D; d += blockDim.x)
    x[(size_t)m * D + d] = tok[(size_t)id * D + d] * sc + pos[(size_t)r * D + d];
}

// ---------------------------------------------------------------------------
// 3) LayerNorm row -> bf16 output
// ---------------------------------------------------------------------------
__global__ void ln_bf16_kernel(const float* __restrict__ x,
                               const float* __restrict__ g,
                               const float* __restrict__ b,
                               __bf16* __restrict__ out, int D) {
  __shared__ float red[256];
  int row = blockIdx.x;
  int tid = threadIdx.x;
  const float* xr = x + (size_t)row * D;

  float s = 0.f;
  for (int d = tid; d < D; d += 256) s += xr[d];
  red[tid] = s; __syncthreads();
  for (int st = 128; st > 0; st >>= 1) {
    if (tid < st) red[tid] += red[tid + st];
    __syncthreads();
  }
  float mean = red[0] / (float)D;
  __syncthreads();

  s = 0.f;
  for (int d = tid; d < D; d += 256) { float t = xr[d] - mean; s += t * t; }
  red[tid] = s; __syncthreads();
  for (int st = 128; st > 0; st >>= 1) {
    if (tid < st) red[tid] += red[tid + st];
    __syncthreads();
  }
  float inv = rsqrtf(red[0] / (float)D + 1e-5f);

  for (int d = tid; d < D; d += 256)
    out[(size_t)row * D + d] = to_bf16((xr[d] - mean) * inv * g[d] + b[d]);
}

// ---------------------------------------------------------------------------
// 4) f32 -> bf16 cast
// ---------------------------------------------------------------------------
__global__ void cast_bf16_kernel(const float* __restrict__ x,
                                 __bf16* __restrict__ y, int n) {
  int i = blockIdx.x * blockDim.x + threadIdx.x;
  if (i < n) y[i] = to_bf16(x[i]);
}

// ---------------------------------------------------------------------------
// 5) Weight pack: fp32 (possibly per-head-blocked) -> bf16 B^T layout [N, K]
//    out[n*K + k] = W[(n/nb)*blockStride + k*ldb + (n%nb)]
//    Tiled LDS transpose: reads coalesced along n, writes coalesced along k.
// ---------------------------------------------------------------------------
__global__ __launch_bounds__(256)
void pack_weight_kernel(const float* __restrict__ W, __bf16* __restrict__ out,
                        int K, int N, int ldb, int nb, long long blockStride) {
  __shared__ __bf16 tile[32][33];
  int k0 = blockIdx.x * 32, n0 = blockIdx.y * 32;
  int tx = threadIdx.x, ty = threadIdx.y;          // 32 x 8
#pragma unroll
  for (int i = ty; i < 32; i += 8) {
    int k = k0 + i, n = n0 + tx;
    float w = 0.f;
    if (k < K && n < N)
      w = W[(size_t)(n / nb) * (size_t)blockStride + (size_t)k * ldb + (n % nb)];
    tile[i][tx] = to_bf16(w);
  }
  __syncthreads();
#pragma unroll
  for (int i = ty; i < 32; i += 8) {
    int n = n0 + i, k = k0 + tx;
    if (k < K && n < N) out[(size_t)n * K + k] = tile[tx][i];
  }
}

// ---------------------------------------------------------------------------
// 6) WMMA GEMM (templated; branch-free inner loop):
//    C[M,N] (fp32 acc) = A_bf16[M,K] @ B
//    WBF16=1: B packed bf16 [N,K], contiguous v16bf fragment loads.
//    WBF16=0: B fp32 [N,ldb] read transposed (token_emb path), float4 loads.
//    OUTMODE: 0 = row-major [M,N]; 1 = scatter [B,H,L,DH] (q/k);
//             2 = scatter transposed [B,H,DH,L] (v -> contiguous PV frags).
//    Each wave computes a 64x16 C tile (4 M-subtiles share one B fragment).
//    Double-buffered K-loop; out-of-range N columns clamped for addressing.
// ---------------------------------------------------------------------------
template <int WBF16, int OUTMODE, int OUTBF16, int GELU>
__global__ __launch_bounds__(256)
void wmma_gemm_kernel(const __bf16* __restrict__ A, int lda,
                      const void* __restrict__ Wp, int ldb,
                      const float* __restrict__ bias,
                      const float* __restrict__ resid,
                      void* __restrict__ Cout,
                      int M, int N, int K,
                      int Lseq, int H, int DH) {
  int lane  = threadIdx.x;               // 0..31
  int waveI = threadIdx.y;               // 0..7
  int tileN = blockIdx.x;
  int tileM4 = blockIdx.y * 8 + waveI;   // group of 4 M-subtiles
  if (tileM4 * 64 >= M) return;

  int half = lane >> 4;                  // 0 or 1
  int lq   = lane & 15;

  // A row pointers for the 4 M-subtiles this wave owns.
  const __bf16* a0p = A + (size_t)(tileM4 * 64 +  0 + lq) * lda + half * 16;
  const __bf16* a1p = A + (size_t)(tileM4 * 64 + 16 + lq) * lda + half * 16;
  const __bf16* a2p = A + (size_t)(tileM4 * 64 + 32 + lq) * lda + half * 16;
  const __bf16* a3p = A + (size_t)(tileM4 * 64 + 48 + lq) * lda + half * 16;

  // Column for B/out; clamp for addressing, guard only stores.
  int ncol = tileN * 16 + lq;
  bool ncolOK = (ncol < N);
  int ncolC = ncolOK ? ncol : (N - 1);
  const __bf16* wb = (const __bf16*)Wp + (size_t)ncolC * K + half * 16;
  const float*  wf = (const float*)Wp + (size_t)ncolC * ldb + half * 16;

  auto loadA = [&](const __bf16* p, int k0) -> v16bf {
    v8bf lo = *(const v8bf*)(p + k0);
    v8bf hi = *(const v8bf*)(p + k0 + 8);
    return __builtin_shufflevector(lo, hi,
        0,1,2,3,4,5,6,7,8,9,10,11,12,13,14,15);
  };
  auto loadB = [&](int k0) -> v16bf {
    if (WBF16) {
      return *(const v16bf*)(wb + k0);
    } else {
      const float* p = wf + k0;
      v16bf r;
#pragma unroll
      for (int e = 0; e < 16; e += 4) {
        float4 w = *(const float4*)(p + e);
        r[e]     = to_bf16(w.x);
        r[e + 1] = to_bf16(w.y);
        r[e + 2] = to_bf16(w.z);
        r[e + 3] = to_bf16(w.w);
      }
      return r;
    }
  };

  v8f acc0 = {}, acc1 = {}, acc2 = {}, acc3 = {};

  // Software pipeline: preload step 0, then load(step+1) before wmma(step).
  v16bf bcur = loadB(0);
  v16bf f0 = loadA(a0p, 0), f1 = loadA(a1p, 0),
        f2 = loadA(a2p, 0), f3 = loadA(a3p, 0);

  int nSteps = K >> 5;
  for (int s = 0; s + 1 < nSteps; ++s) {
    int kn = (s + 1) << 5;
    v16bf bn = loadB(kn);
    v16bf n0 = loadA(a0p, kn), n1 = loadA(a1p, kn),
          n2 = loadA(a2p, kn), n3 = loadA(a3p, kn);
    acc0 = __builtin_amdgcn_wmma_f32_16x16x32_bf16(false, f0, false, bcur, (short)0, acc0, false, false);
    acc1 = __builtin_amdgcn_wmma_f32_16x16x32_bf16(false, f1, false, bcur, (short)0, acc1, false, false);
    acc2 = __builtin_amdgcn_wmma_f32_16x16x32_bf16(false, f2, false, bcur, (short)0, acc2, false, false);
    acc3 = __builtin_amdgcn_wmma_f32_16x16x32_bf16(false, f3, false, bcur, (short)0, acc3, false, false);
    bcur = bn; f0 = n0; f1 = n1; f2 = n2; f3 = n3;
  }
  acc0 = __builtin_amdgcn_wmma_f32_16x16x32_bf16(false, f0, false, bcur, (short)0, acc0, false, false);
  acc1 = __builtin_amdgcn_wmma_f32_16x16x32_bf16(false, f1, false, bcur, (short)0, acc1, false, false);
  acc2 = __builtin_amdgcn_wmma_f32_16x16x32_bf16(false, f2, false, bcur, (short)0, acc2, false, false);
  acc3 = __builtin_amdgcn_wmma_f32_16x16x32_bf16(false, f3, false, bcur, (short)0, acc3, false, false);

  // ---- Epilogue. C layout: lane -> col = lq, vgpr r -> row = r + half*8.
  float bi = (bias != nullptr) ? bias[ncolC] : 0.f;
  v8f accs[4] = {acc0, acc1, acc2, acc3};
#pragma unroll
  for (int t = 0; t < 4; ++t) {
#pragma unroll
    for (int r = 0; r < 8; ++r) {
      if (!ncolOK) continue;
      int row = tileM4 * 64 + t * 16 + r + half * 8;
      int col = ncol;
      float v = accs[t][r] + bi;
      if (GELU) v = 0.5f * v * (1.f + erff(v * 0.7071067811865475f));
      size_t oidx;
      if (OUTMODE == 1) {        // scatter to [B, H, Lseq, DH]
        int bI = row / Lseq, l = row % Lseq;
        int h = col / DH,   dh = col % DH;
        oidx = (((size_t)bI * H + h) * Lseq + l) * DH + dh;
      } else if (OUTMODE == 2) { // scatter transposed to [B, H, DH, Lseq]
        int bI = row / Lseq, l = row % Lseq;
        int h = col / DH,   dh = col % DH;
        oidx = (((size_t)bI * H + h) * DH + dh) * Lseq + l;
      } else {
        oidx = (size_t)row * N + col;
      }
      if (resid != nullptr) v += resid[oidx];
      if (OUTBF16) ((__bf16*)Cout)[oidx] = to_bf16(v);
      else         ((float*)Cout)[oidx]  = v;
    }
  }
}

// ---------------------------------------------------------------------------
// 7) Flash attention (one wave per (b, head, 16-query tile)).
//    q,k: bf16 [B,H,L,DH]; v: bf16 TRANSPOSED [B,H,DH,L]; seg: int [B,L].
//    Writes o (bf16) into hbuf[B*L, D] at columns head*DH .. head*DH+63.
//    logits = Q K^T + mask (NO 1/sqrt(dh) scaling, per reference).
// ---------------------------------------------------------------------------
__global__ __launch_bounds__(32)
void flash_attn_kernel(const __bf16* __restrict__ Q,
                       const __bf16* __restrict__ Km,
                       const __bf16* __restrict__ Vt,
                       const int* __restrict__ seg,
                       __bf16* __restrict__ hbuf,
                       int L, int H, int DH, int D) {
  __shared__ __bf16 pbuf[16 * 32];   // P tile, 16 rows x 32 keys

  int lane  = threadIdx.x;
  int half  = lane >> 4;
  int lq    = lane & 15;
  int qTile = blockIdx.x;
  int head  = blockIdx.y;
  int bI    = blockIdx.z;
  int qBase = qTile * 16;

  const size_t bh = ((size_t)bI * H + head) * L;
  const __bf16* Qp = Q  + bh * DH;
  const __bf16* Kp = Km + bh * DH;
  const __bf16* Vp = Vt + bh * DH;     // [DH, L] per (b,h): same total offset
  const int*    sg = seg + (size_t)bI * L;

  // Preload Q A-fragments for K-chunks [0,32) and [32,64).
  const __bf16* qr = Qp + (size_t)(qBase + lq) * DH;
  v16bf aq[2];
#pragma unroll
  for (int c = 0; c < 2; ++c) {
    const __bf16* ap = qr + c * 32 + half * 16;
    v8bf lo = *(const v8bf*)(ap);
    v8bf hi = *(const v8bf*)(ap + 8);
    aq[c] = __builtin_shufflevector(lo, hi, 0,1,2,3,4,5,6,7,8,9,10,11,12,13,14,15);
  }

  // Per-lane query-row segments (rows r + half*8 of this tile).
  int   segq[8];
  float m_run[8], s_run[8];
#pragma unroll
  for (int r = 0; r < 8; ++r) {
    segq[r]  = sg[qBase + r + half * 8];
    m_run[r] = -INFINITY;
    s_run[r] = 0.f;
  }
  v8f oacc[4] = {};   // O[16 x 64], 4 column chunks of 16

  int nBlk = qTile / 2 + 1;   // 32-key blocks covering causal range
  for (int kb = 0; kb < nBlk; ++kb) {
    int keyBase = kb * 32;

    // ---- logits for two 16-key tiles: QK^T over DH=64 (2 chunks of 32)
    v8f c0 = {}, c1 = {};
#pragma unroll
    for (int c = 0; c < 2; ++c) {
      // B element (kk, n) = K[keyBase + n][c*32 + kk]; contiguous 16 bf16.
      v16bf bk0 = *(const v16bf*)(Kp + (size_t)(keyBase + lq)      * DH + c * 32 + half * 16);
      v16bf bk1 = *(const v16bf*)(Kp + (size_t)(keyBase + 16 + lq) * DH + c * 32 + half * 16);
      c0 = __builtin_amdgcn_wmma_f32_16x16x32_bf16(false, aq[c], false, bk0, (short)0, c0, false, false);
      c1 = __builtin_amdgcn_wmma_f32_16x16x32_bf16(false, aq[c], false, bk1, (short)0, c1, false, false);
    }

    // ---- mask + online softmax stats
    int key0 = keyBase + lq, key1 = key0 + 16;
    int sk0 = sg[key0], sk1 = sg[key1];
    float p0[8], p1[8];
#pragma unroll
    for (int r = 0; r < 8; ++r) {
      int row = qBase + r + half * 8;
      float l0 = (key0 <= row && sk0 == segq[r]) ? c0[r] : -INFINITY;
      float l1 = (key1 <= row && sk1 == segq[r]) ? c1[r] : -INFINITY;
      float bm = fmaxf(l0, l1);
      // row-max across the 16 lanes of this half-wave
      bm = fmaxf(bm, __shfl_xor(bm, 1, 32));
      bm = fmaxf(bm, __shfl_xor(bm, 2, 32));
      bm = fmaxf(bm, __shfl_xor(bm, 4, 32));
      bm = fmaxf(bm, __shfl_xor(bm, 8, 32));
      float newm = fmaxf(m_run[r], bm);
      float alpha = (m_run[r] > -INFINITY) ? expf(m_run[r] - newm) : 0.f;
      float e0 = (l0 > -INFINITY) ? expf(l0 - newm) : 0.f;
      float e1 = (l1 > -INFINITY) ? expf(l1 - newm) : 0.f;
      float rs = e0 + e1;
      rs += __shfl_xor(rs, 1, 32);
      rs += __shfl_xor(rs, 2, 32);
      rs += __shfl_xor(rs, 4, 32);
      rs += __shfl_xor(rs, 8, 32);
      s_run[r] = s_run[r] * alpha + rs;
      m_run[r] = newm;
      p0[r] = e0; p1[r] = e1;
#pragma unroll
      for (int cc = 0; cc < 4; ++cc) oacc[cc][r] *= alpha;
    }

    // ---- stage P (16x32) through LDS to reshape C-layout -> A-fragment
#pragma unroll
    for (int r = 0; r < 8; ++r) {
      int row = r + half * 8;
      pbuf[row * 32 + lq]      = to_bf16(p0[r]);
      pbuf[row * 32 + 16 + lq] = to_bf16(p1[r]);
    }
    asm volatile("s_wait_dscnt 0" ::: "memory");

    const __bf16* pp = &pbuf[lq * 32 + half * 16];
    v8bf plo = *(const v8bf*)(pp);
    v8bf phi = *(const v8bf*)(pp + 8);
    v16bf pa = __builtin_shufflevector(plo, phi, 0,1,2,3,4,5,6,7,8,9,10,11,12,13,14,15);

    // ---- O += P @ V  (K=32 keys, 4 column chunks)
    //      B element (kk, n) = Vt[(cc*16 + n)*L + keyBase + kk]:
    //      contiguous 16 bf16 per lane (one 32B vector load per chunk).
#pragma unroll
    for (int cc = 0; cc < 4; ++cc) {
      v16bf bv = *(const v16bf*)(Vp + (size_t)(cc * 16 + lq) * L +
                                 keyBase + half * 16);
      oacc[cc] = __builtin_amdgcn_wmma_f32_16x16x32_bf16(
          false, pa, false, bv, (short)0, oacc[cc], false, false);
    }
  }

  // ---- normalize + write o into hbuf at columns head*DH + ...
#pragma unroll
  for (int cc = 0; cc < 4; ++cc) {
#pragma unroll
    for (int r = 0; r < 8; ++r) {
      int row = qBase + r + half * 8;
      int col = head * DH + cc * 16 + lq;
      float v = oacc[cc][r] / s_run[r];
      hbuf[((size_t)bI * L + row) * D + col] = to_bf16(v);
    }
  }
}

// ---------------------------------------------------------------------------
// Host-side orchestration
// ---------------------------------------------------------------------------
extern "C" void kernel_launch(void* const* d_in, const int* in_sizes, int n_in,
                              void* d_out, int out_size, void* d_ws, size_t ws_size,
                              hipStream_t stream) {
  (void)in_sizes; (void)n_in; (void)out_size; (void)ws_size;
  const int B = 2, L = 1024, D = 768, H = 12, DH = 64, NL = 4, FF = 3072, V = 50257;
  const int M = B * L;

  const int*   ids  = (const int*)  d_in[0];
  const float* tok  = (const float*)d_in[1];
  const float* pos  = (const float*)d_in[2];
  const float* ln1g = (const float*)d_in[3];
  const float* ln1b = (const float*)d_in[4];
  const float* Wq   = (const float*)d_in[5];
  const float* bq   = (const float*)d_in[6];
  const float* Wk   = (const float*)d_in[7];
  const float* bk   = (const float*)d_in[8];
  const float* Wv   = (const float*)d_in[9];
  const float* bv   = (const float*)d_in[10];
  const float* Wo   = (const float*)d_in[11];
  const float* bo   = (const float*)d_in[12];
  const float* ln2g = (const float*)d_in[13];
  const float* ln2b = (const float*)d_in[14];
  const float* W1   = (const float*)d_in[15];
  const float* b1   = (const float*)d_in[16];
  const float* W2   = (const float*)d_in[17];
  const float* b2   = (const float*)d_in[18];
  float* out = (float*)d_out;

  char* ws = (char*)d_ws;
  size_t off = 0;
  auto alloc = [&](size_t bytes) -> void* {
    void* p = ws + off;
    off = (off + bytes + 255) & ~(size_t)255;
    return p;
  };
  int*    seg = (int*)   alloc((size_t)M * 4);
  int*    rel = (int*)   alloc((size_t)M * 4);
  float*  x   = (float*) alloc((size_t)M * D * 4);
  __bf16* hb  = (__bf16*)alloc((size_t)M * D * 2);
  __bf16* qB  = (__bf16*)alloc((size_t)M * D * 2);
  __bf16* kB  = (__bf16*)alloc((size_t)M * D * 2);
  __bf16* vT  = (__bf16*)alloc((size_t)M * D * 2);   // [B,H,DH,L]
  __bf16* ffb = (__bf16*)alloc((size_t)M * FF * 2);
  __bf16* pw  = (__bf16*)alloc((size_t)D * FF * 2);  // packed weight buffer (reused)

  dim3 gblk(32, 8);                       // 8 waves; each wave: 64x16 C tile
  dim3 gemmD(D / 16,  M / 512);           // N=768
  dim3 gemmF(FF / 16, M / 512);           // N=3072
  dim3 gemmV((V + 15) / 16, M / 512);     // N=50257 (store-guarded)
  dim3 attnG(L / 16, H, B);
  dim3 pblk(32, 8);
  dim3 packDD((D + 31) / 32,  (D + 31) / 32);   // K=D,  N=D
  dim3 packDF((D + 31) / 32,  (FF + 31) / 32);  // K=D,  N=FF
  dim3 packFD((FF + 31) / 32, (D + 31) / 32);   // K=FF, N=D

  seg_scan_kernel<<<B, 32, 0, stream>>>(ids, seg, rel, L);
  embed_kernel<<<M, 256, 0, stream>>>(ids, rel, tok, pos, x, D);

  const long long hs = (long long)D * DH;   // per-head weight block stride

  for (int i = 0; i < NL; ++i) {
    ln_bf16_kernel<<<M, 256, 0, stream>>>(x, ln1g + i * D, ln1b + i * D, hb, D);

    // --- QKV (pack per-head-blocked weight, then GEMM with bf16 B) ---
    pack_weight_kernel<<<packDD, pblk, 0, stream>>>(
        Wq + (size_t)i * H * D * DH, pw, D, D, DH, DH, hs);
    wmma_gemm_kernel<1, 1, 1, 0><<<gemmD, gblk, 0, stream>>>(
        hb, D, pw, D, bq + (size_t)i * H * DH, nullptr, qB,
        M, D, D, L, H, DH);
    pack_weight_kernel<<<packDD, pblk, 0, stream>>>(
        Wk + (size_t)i * H * D * DH, pw, D, D, DH, DH, hs);
    wmma_gemm_kernel<1, 1, 1, 0><<<gemmD, gblk, 0, stream>>>(
        hb, D, pw, D, bk + (size_t)i * H * DH, nullptr, kB,
        M, D, D, L, H, DH);
    pack_weight_kernel<<<packDD, pblk, 0, stream>>>(
        Wv + (size_t)i * H * D * DH, pw, D, D, DH, DH, hs);
    wmma_gemm_kernel<1, 2, 1, 0><<<gemmD, gblk, 0, stream>>>(   // V transposed
        hb, D, pw, D, bv + (size_t)i * H * DH, nullptr, vT,
        M, D, D, L, H, DH);

    flash_attn_kernel<<<attnG, 32, 0, stream>>>(qB, kB, vT, seg, hb, L, H, DH, D);

    // --- output projection (+residual) ---
    pack_weight_kernel<<<packDD, pblk, 0, stream>>>(
        Wo + (size_t)i * D * D, pw, D, D, D, D, 0);
    wmma_gemm_kernel<1, 0, 0, 0><<<gemmD, gblk, 0, stream>>>(
        hb, D, pw, D, bo + (size_t)i * D, x, x,
        M, D, D, L, H, DH);

    ln_bf16_kernel<<<M, 256, 0, stream>>>(x, ln2g + i * D, ln2b + i * D, hb, D);

    // --- MLP up (+GELU, bf16 out) ---
    pack_weight_kernel<<<packDF, pblk, 0, stream>>>(
        W1 + (size_t)i * D * FF, pw, D, FF, FF, FF, 0);
    wmma_gemm_kernel<1, 0, 1, 1><<<gemmF, gblk, 0, stream>>>(
        hb, D, pw, D, b1 + (size_t)i * FF, nullptr, ffb,
        M, FF, D, L, H, DH);

    // --- MLP down (+residual) ---
    pack_weight_kernel<<<packFD, pblk, 0, stream>>>(
        W2 + (size_t)i * FF * D, pw, FF, D, D, D, 0);
    wmma_gemm_kernel<1, 0, 0, 0><<<gemmD, gblk, 0, stream>>>(
        ffb, FF, pw, FF, b2 + (size_t)i * D, x, x,
        M, D, FF, L, H, DH);
  }

  // Final tied decoder: out[M,V] = x_bf16 @ token_emb^T (fp32 B path)
  cast_bf16_kernel<<<(M * D + 255) / 256, 256, 0, stream>>>(x, hb, M * D);
  wmma_gemm_kernel<0, 0, 0, 0><<<gemmV, gblk, 0, stream>>>(
      hb, D, tok, D, nullptr, nullptr, out,
      M, V, D, L, H, DH);
}
